// Network_39041252721132
// MI455X (gfx1250) — matrix operand order repaired and checked by reading, not compiled, not run on hardware
//
#include <hip/hip_runtime.h>

#define N_DIM    4096
#define NLAYERS  2048
#define N_HIST   64
#define NBLK     256        // one WG per 16-row tile of the output vector
#define TPB      256        // 8 wave32 waves per WG
#define WAVES    8
#define KSLICE   (N_DIM / WAVES)   // 512 columns of W per wave
#define KSTEPS   (KSLICE / 32)     // 16 WMMA k-steps per wave per layer

typedef __attribute__((ext_vector_type(16))) __bf16        v16bf;
typedef __attribute__((ext_vector_type(8)))  float         v8f;
typedef __attribute__((ext_vector_type(8)))  unsigned int  v8u;

// fp32 -> bf16, round-to-nearest-even
__device__ __forceinline__ unsigned short f2bf(float f) {
    unsigned u = __builtin_bit_cast(unsigned, f);
    unsigned r = (u + 0x7FFFu + ((u >> 16) & 1u)) >> 16;
    return (unsigned short)r;
}

// ---------------------------------------------------------------------------
// Phase 1: one-shot W fp32 -> bf16 conversion into workspace (L2-resident)
// ---------------------------------------------------------------------------
__global__ void convert_w_bf16(const float* __restrict__ W,
                               unsigned short* __restrict__ Wb, int n) {
    int i = blockIdx.x * blockDim.x + threadIdx.x;
    int stride = gridDim.x * blockDim.x;
    for (; i < n; i += stride) Wb[i] = f2bf(W[i]);
}

// ---------------------------------------------------------------------------
// Phase 2: persistent recurrence kernel. 256 WGs cooperate on each GEMV,
// synchronized per layer by a monotonic device-wide atomic barrier.
// ---------------------------------------------------------------------------
__global__ void __launch_bounds__(TPB)
reservoir_persistent(const float* __restrict__ init_state,
                     const float* __restrict__ biases,
                     const unsigned short* __restrict__ Wb,
                     const float* __restrict__ wscale_p,
                     const float* __restrict__ bscale_p,
                     float* __restrict__ sbuf0,
                     float* __restrict__ sbuf1,
                     unsigned* __restrict__ barrier_cnt,
                     float* __restrict__ out)
{
    __shared__ unsigned short s_bf[N_DIM];     // current state, bf16 (8 KB)
    __shared__ float          partial[WAVES][16];

    const int tid   = threadIdx.x;
    const int wave  = tid >> 5;
    const int lane  = tid & 31;
    const int rowbase = blockIdx.x * 16;            // 16-row tile of output
    const int row     = rowbase + (lane & 15);
    const int kwave   = wave * KSLICE;              // this wave's K window

    const float wscale     = wscale_p[0];
    const float bscale     = bscale_p[0];
    const float inv_sqrt_n = 0.015625f;             // 1/sqrt(4096)

    // ISA A-layout (16-bit 16x32): lanes 0-15 hold K=0..7 / 16..23,
    // lanes 16-31 hold K=8..15 / 24..31 of their row.
    const int off8  = (lane < 16) ? 0 : 8;
    // B-layout: same K order as A (lanes 0-15 K=0..7/16..23, lanes 16-31 K=8..15/24..31).
    const int off16 = (lane < 16) ? 0 : 16;

    const unsigned short* wrow = Wb + (size_t)row * N_DIM;
    const unsigned int*   sb32 = (const unsigned int*)s_bf;
    float* sbufs[2] = { sbuf0, sbuf1 };

    for (int l = 0; l < NLAYERS; ++l) {
        const float* s_in  = (l == 0) ? init_state : sbufs[(l & 1) ^ 1];
        float*       s_out = sbufs[l & 1];

        // Stage this wave's 512-float state slice into LDS as bf16.
        // Agent-scope loads bypass stale WGP-cache lines from 2 layers ago.
        #pragma unroll
        for (int i = 0; i < 16; ++i) {
            int idx = kwave + i * 32 + lane;
            float v = __hip_atomic_load((float*)&s_in[idx],
                                        __ATOMIC_RELAXED, __HIP_MEMORY_SCOPE_AGENT);
            s_bf[idx] = f2bf(v);
        }
        __syncthreads();

        v8f c = {};
        #pragma unroll 4
        for (int t = 0; t < KSTEPS; ++t) {
            const int K0 = kwave + t * 32;

            // A tile: W rows [rowbase, rowbase+16), cols [K0, K0+32)
            const unsigned short* ap = wrow + K0 + off8;
            uint4 lo = *(const uint4*)(ap);        // 8 bf16
            uint4 hi = *(const uint4*)(ap + 16);   // 8 bf16, +16 elements
            v8u au;
            au[0] = lo.x; au[1] = lo.y; au[2] = lo.z; au[3] = lo.w;
            au[4] = hi.x; au[5] = hi.y; au[6] = hi.z; au[7] = hi.w;

            // B tile: state chunk s[K0..K0+32) replicated into all 16 columns
            const unsigned base = (unsigned)(K0 + off8) >> 1;    // dword index
            v8u bu;
            #pragma unroll
            for (int r = 0; r < 8; ++r) bu[r] = sb32[base + (r < 4 ? r : r + 4)];

            v16bf a = __builtin_bit_cast(v16bf, au);
            v16bf b = __builtin_bit_cast(v16bf, bu);
            c = __builtin_amdgcn_wmma_f32_16x16x32_bf16(
                    false, a, false, b, (short)0, c, false, false);
        }

        // Column 0 of D carries the 16 dot products:
        // lane 0 VGPR r -> M=r, lane 16 VGPR r -> M=8+r.
        if (lane == 0) {
            #pragma unroll
            for (int r = 0; r < 8; ++r) partial[wave][r] = c[r];
        } else if (lane == 16) {
            #pragma unroll
            for (int r = 0; r < 8; ++r) partial[wave][8 + r] = c[r];
        }
        __syncthreads();

        if (tid < 16) {
            float dot = 0.f;
            #pragma unroll
            for (int wv = 0; wv < WAVES; ++wv) dot += partial[wv][tid];
            const int orow = rowbase + tid;
            float bval = biases[(size_t)l * N_DIM + orow];
            float nv = erff(wscale * dot + bscale * bval) * inv_sqrt_n;
            __hip_atomic_store(&s_out[orow], nv,
                               __ATOMIC_RELAXED, __HIP_MEMORY_SCOPE_AGENT);
            if (l >= NLAYERS - N_HIST)
                out[(size_t)(l - (NLAYERS - N_HIST)) * N_DIM + orow] = nv;
        }

        // Device-wide monotonic barrier (release/acquire, agent scope).
        __syncthreads();
        if (tid == 0) {
            __hip_atomic_fetch_add(barrier_cnt, 1u,
                                   __ATOMIC_RELEASE, __HIP_MEMORY_SCOPE_AGENT);
            const unsigned target = (unsigned)(l + 1) * NBLK;
            while (__hip_atomic_load(barrier_cnt,
                                     __ATOMIC_ACQUIRE, __HIP_MEMORY_SCOPE_AGENT) < target)
                __builtin_amdgcn_s_sleep(2);
        }
        __syncthreads();
    }
}

extern "C" void kernel_launch(void* const* d_in, const int* in_sizes, int n_in,
                              void* d_out, int out_size, void* d_ws, size_t ws_size,
                              hipStream_t stream) {
    const float* init   = (const float*)d_in[0];   // [4096]
    const float* biases = (const float*)d_in[1];   // [2048, 4096]
    const float* W      = (const float*)d_in[2];   // [4096, 4096]
    const float* wscale = (const float*)d_in[3];   // [1]
    const float* bscale = (const float*)d_in[4];   // [1]
    (void)in_sizes; (void)n_in; (void)out_size;

    // Workspace layout: [0,256) barrier counter, then 2 state buffers (32 KB),
    // then bf16 W (32 MB, lives in the 192 MB L2 across all 2048 layers).
    char* ws = (char*)d_ws;
    unsigned* cnt        = (unsigned*)ws;
    float* sbuf0         = (float*)(ws + 256);
    float* sbuf1         = sbuf0 + N_DIM;
    unsigned short* Wb   = (unsigned short*)(ws + 256 + 2 * N_DIM * sizeof(float));
    (void)ws_size;

    // Reset barrier each call so graph replays are deterministic.
    hipMemsetAsync(cnt, 0, 256, stream);

    convert_w_bf16<<<2048, 256, 0, stream>>>(W, Wb, N_DIM * N_DIM);

    reservoir_persistent<<<NBLK, TPB, 0, stream>>>(
        init, biases, Wb, wscale, bscale, sbuf0, sbuf1, cnt, (float*)d_out);
}
